// MultiHeadAttention_11501922419041
// MI455X (gfx1250) — matrix operand order repaired and checked by reading, not compiled, not run on hardware
//
#include <hip/hip_runtime.h>
#include <hip/hip_bf16.h>

// ---------------------------------------------------------------------------
// MultiHeadAttention for MI455X (gfx1250), wave32 + WMMA f32_16x16x32_f16
// + async global->LDS staging (ASYNCcnt), double-buffered & software-pipelined.
// B=2, S=2048, E=768, H=12, Dk=64.
// ---------------------------------------------------------------------------

typedef __attribute__((ext_vector_type(16))) _Float16 v16h;
typedef __attribute__((ext_vector_type(8)))  _Float16 v8h;
typedef __attribute__((ext_vector_type(8)))  float    v8f;

#define BB 2
#define SS 2048
#define EE 768
#define HH 12
#define DK 64
#define MM (BB * SS)   // 4096

__device__ __forceinline__ v16h cat8(v8h lo, v8h hi) {
    return __builtin_shufflevector(lo, hi, 0, 1, 2, 3, 4, 5, 6, 7,
                                           8, 9, 10, 11, 12, 13, 14, 15);
}

__device__ __forceinline__ v8f wmma_f16(v16h a, v16h b, v8f c) {
    return __builtin_amdgcn_wmma_f32_16x16x32_f16(
        /*neg_a=*/false, a, /*neg_b=*/false, b,
        /*c_mod=*/(short)0, c, /*reuse_a=*/false, /*reuse_b=*/false);
}

// Async copy of 16 bytes global -> LDS, tracked by ASYNCcnt.
// LDS aperture: low 32 bits of the flat address are the LDS byte address.
__device__ __forceinline__ void async_ldg_b128(void* lds_dst, const void* gsrc) {
    const unsigned int  l = (unsigned int)(unsigned long long)lds_dst;
    const unsigned long long g = (unsigned long long)gsrc;
    asm volatile("global_load_async_to_lds_b128 %0, %1, off"
                 :: "v"(l), "v"(g)
                 : "memory");
}

__device__ __forceinline__ void wait_async0() {
    asm volatile("s_wait_asynccnt 0" ::: "memory");
}

// ---------------------------------------------------------------------------
// Generic GEMM: C[M=4096, N=768] = A[4096, 768] * B'[768, 768]
//   BMODE 0: B'[e,n] = Bw[e*768 + n]            (plain row-major, fp32)
//   BMODE 1: B'[e,n] = Bw[(n/64)*768*64 + e*64 + (n%64)]  (per-head [H,E,Dk])
//   A_F32:  A fp32 (register prefetch + convert) else f16 (async-to-LDS)
//   OUT_F32: store fp32, else f16
// Block: 256 threads (8 waves). Block tile 128x64; wave tile 16x64; K step 32.
// Double-buffered LDS, one barrier per K-step.
// ---------------------------------------------------------------------------
template <int BMODE, bool A_F32, bool OUT_F32>
__global__ __launch_bounds__(256) void gemm_wmma_kernel(
    const void* __restrict__ Aptr, const float* __restrict__ Bw,
    void* __restrict__ Optr) {
    constexpr int K   = EE;
    constexpr int N   = EE;
    constexpr int LDA = 40;  // 32 + 8 halves pad
    constexpr int LDB = 40;

    __shared__ _Float16 As[2][128 * LDA];   // [row][k]
    __shared__ _Float16 Bt[2][64 * LDB];    // [n][k] (transposed tile)

    const int tid  = threadIdx.x;
    const int w    = tid >> 5;
    const int lane = tid & 31;
    const int lcol = lane & 15;
    const int kbA  = (lane & 16) ? 16 : 0;   // A-frag K base (contiguous)
    const int kbB  = (lane & 16) ? 16 : 0;   // B-frag K base
    const int hoff = (lane & 16) ? 8 : 0;    // C-frag row offset

    const int m0 = blockIdx.y * 128;
    const int n0 = blockIdx.x * 64;

    v8f acc[4] = {};

    const float*    Af = (const float*)Aptr;
    const _Float16* Ah = (const _Float16*)Aptr;

    // staging assignments
    const int arow = tid >> 1;          // 0..127
    const int acb  = (tid & 1) * 16;    // 0 or 16
    const int brow = tid >> 3;          // 0..31  (k within tile)
    const int bcb  = (tid & 7) * 8;     // 0..56  (n within tile)

    float areg[16];
    float breg[8];

    auto loadA = [&](int kk0) {
        if constexpr (A_F32) {
            const float4* s =
                (const float4*)(Af + (size_t)(m0 + arow) * K + kk0 + acb);
#pragma unroll
            for (int i = 0; i < 4; i++) {
                float4 f = s[i];
                areg[4 * i + 0] = f.x; areg[4 * i + 1] = f.y;
                areg[4 * i + 2] = f.z; areg[4 * i + 3] = f.w;
            }
        }
    };
    auto storeA = [&](int buf) {
        if constexpr (A_F32) {
            _Float16* d = &As[buf][arow * LDA + acb];
#pragma unroll
            for (int i = 0; i < 16; i++) d[i] = (_Float16)areg[i];
        }
    };
    auto asyncA = [&](int buf, int kk0) {
        const _Float16* src = Ah + (size_t)(m0 + arow) * K + kk0 + acb;
        async_ldg_b128(&As[buf][arow * LDA + acb],     src);
        async_ldg_b128(&As[buf][arow * LDA + acb + 8], src + 8);
    };
    auto loadB = [&](int kk0) {
        const int e = kk0 + brow;
        const float* bp = (BMODE == 1)
            ? Bw + (size_t)((n0 + bcb) >> 6) * (K * 64) + (size_t)e * 64 +
                  ((n0 + bcb) & 63)
            : Bw + (size_t)e * N + n0 + bcb;
        float4 f0 = ((const float4*)bp)[0];
        float4 f1 = ((const float4*)bp)[1];
        breg[0] = f0.x; breg[1] = f0.y; breg[2] = f0.z; breg[3] = f0.w;
        breg[4] = f1.x; breg[5] = f1.y; breg[6] = f1.z; breg[7] = f1.w;
    };
    auto storeB = [&](int buf) {
#pragma unroll
        for (int i = 0; i < 8; i++)
            Bt[buf][(bcb + i) * LDB + brow] = (_Float16)breg[i];
    };

    // ---- prologue: stage tile 0 into buffer 0 ----
    if constexpr (A_F32) { loadA(0); storeA(0); } else { asyncA(0, 0); }
    loadB(0); storeB(0);
    if constexpr (!A_F32) wait_async0();
    __syncthreads();

    for (int kk0 = 0; kk0 < K; kk0 += 32) {
        const int  buf      = (kk0 >> 5) & 1;
        const bool has_next = (kk0 + 32 < K);

        // issue next tile's staging (overlaps with WMMAs below)
        if (has_next) {
            if constexpr (A_F32) loadA(kk0 + 32);
            else                 asyncA(buf ^ 1, kk0 + 32);
            loadB(kk0 + 32);
            if (kk0 + 64 < K) {  // warm L2 one more tile ahead
                const char* pf = A_F32
                    ? (const char*)(Af + (size_t)(m0 + arow) * K + kk0 + 64)
                    : (const char*)(Ah + (size_t)(m0 + arow) * K + kk0 + 64);
                __builtin_prefetch(pf, 0, 1);
            }
        }

        // ---- fragments (all loads first), then back-to-back WMMAs ----
        const _Float16* ap = &As[buf][(w * 16 + lcol) * LDA];
        v16h afrag = cat8(*(const v8h*)(ap + kbA), *(const v8h*)(ap + kbA + 8));
        v16h bfr[4];
#pragma unroll
        for (int nt = 0; nt < 4; nt++) {
            const _Float16* bp = &Bt[buf][(nt * 16 + lcol) * LDB + kbB];
            bfr[nt] = cat8(*(const v8h*)bp, *(const v8h*)(bp + 8));
        }
#pragma unroll
        for (int nt = 0; nt < 4; nt++)
            acc[nt] = wmma_f16(afrag, bfr[nt], acc[nt]);

        // write next tile into the idle buffer; publish with one barrier
        if (has_next) {
            if constexpr (A_F32) storeA(buf ^ 1);
            storeB(buf ^ 1);
            if constexpr (!A_F32) wait_async0();
        }
        __syncthreads();
    }

    // ---- epilogue ----
    const int row_base = m0 + w * 16 + hoff;
#pragma unroll
    for (int nt = 0; nt < 4; nt++) {
#pragma unroll
        for (int r = 0; r < 8; r++) {
            const int row = row_base + r;
            const int col = n0 + nt * 16 + lcol;
            if constexpr (OUT_F32)
                ((float*)Optr)[(size_t)row * N + col] = acc[nt][r];
            else
                ((_Float16*)Optr)[(size_t)row * N + col] = (_Float16)acc[nt][r];
        }
    }
}

// ---------------------------------------------------------------------------
// Flash attention. Grid: (S/128, H, B). Block: 256 threads = 8 waves.
// Each wave: 16 Q rows, full Dk=64. Keys processed 32 at a time.
// K tile: async global->LDS copy, double buffered (overlaps WMMA compute).
// V tile: register-prefetched one tile ahead, transposed into LDS.
// One block barrier per kv-iteration.
// ---------------------------------------------------------------------------
__global__ __launch_bounds__(256) void attn_wmma_kernel(
    const _Float16* __restrict__ Qh, const _Float16* __restrict__ Kh,
    const _Float16* __restrict__ Vh, const int* __restrict__ mask,
    _Float16* __restrict__ AO) {
    constexpr int LDK = 72;  // K tile [key 32][dk 64] + pad
    constexpr int LDV = 40;  // V^T tile [dk 64][key 32] + pad
    constexpr int LDP = 40;  // P tile per wave [row 16][key 32] + pad

    __shared__ _Float16 Kt[2][32 * LDK];
    __shared__ _Float16 Vt[2][64 * LDV];
    __shared__ _Float16 Pl[8 * 16 * LDP];   // wave-private rows; no barrier

    const int tid  = threadIdx.x;
    const int w    = tid >> 5;
    const int lane = tid & 31;
    const int lcol = lane & 15;
    const int kbA  = (lane & 16) ? 16 : 0;
    const int kbB  = (lane & 16) ? 16 : 0;
    const int hoff = (lane & 16) ? 8 : 0;

    const int b  = blockIdx.z;
    const int h  = blockIdx.y;
    const int q0 = blockIdx.x * 128;

    const _Float16* qb = Qh + ((size_t)b * SS + q0) * EE + h * DK;
    const _Float16* kb = Kh + ((size_t)b * SS) * EE + h * DK;
    const _Float16* vb = Vh + ((size_t)b * SS) * EE + h * DK;
    const int*      mb = mask + (size_t)b * SS * SS;

    // Q A-fragments (chunk c covers dk = 32c..32c+31), kept in VGPRs
    v16h qa[2];
#pragma unroll
    for (int c = 0; c < 2; c++) {
        const _Float16* qr = qb + (size_t)(w * 16 + lcol) * EE + c * 32;
        qa[c] = cat8(*(const v8h*)(qr + kbA), *(const v8h*)(qr + kbA + 8));
    }

    float mi[8], li[8];
#pragma unroll
    for (int r = 0; r < 8; r++) { mi[r] = -1e30f; li[r] = 0.0f; }
    v8f acc[4] = {};

    const int key = tid >> 3;   // 0..31
    const int grp = tid & 7;    // 0..7 (8-half group within dk)

    // ---- prologue: tile 0 ----
    async_ldg_b128(&Kt[0][key * LDK + grp * 8],
                   kb + (size_t)key * EE + grp * 8);
    v8h vreg = *(const v8h*)(vb + (size_t)key * EE + grp * 8);

    for (int it = 0; it < SS / 32; ++it) {
        const int kv0 = it * 32;
        const int buf = it & 1;

        // K tile (this wave's portion) has landed; scatter prefetched V tile.
        wait_async0();
#pragma unroll
        for (int i = 0; i < 8; i++) Vt[buf][(grp * 8 + i) * LDV + key] = vreg[i];
        __syncthreads();   // publish K/V tile `it` to all waves

        // Issue next tile's staging: overlaps with compute below.
        if (it + 1 < SS / 32) {
            const size_t nxt = (size_t)(kv0 + 32 + key) * EE + grp * 8;
            async_ldg_b128(&Kt[buf ^ 1][key * LDK + grp * 8], kb + nxt);
            vreg = *(const v8h*)(vb + nxt);
        }

        // ---- Z = Q K^T: load all 4 K-fragments, then 4 WMMAs ----
        v16h kf[4];
#pragma unroll
        for (int ct = 0; ct < 2; ct++) {
#pragma unroll
            for (int c = 0; c < 2; c++) {
                const _Float16* bp =
                    &Kt[buf][(ct * 16 + lcol) * LDK + c * 32 + kbB];
                kf[ct * 2 + c] = cat8(*(const v8h*)bp, *(const v8h*)(bp + 8));
            }
        }
        v8f z[2] = {};
        z[0] = wmma_f16(qa[0], kf[0], z[0]);
        z[1] = wmma_f16(qa[0], kf[2], z[1]);
        z[0] = wmma_f16(qa[1], kf[1], z[0]);
        z[1] = wmma_f16(qa[1], kf[3], z[1]);

        // ---- scale + mask + online softmax (row = r + hoff, col = lcol) ----
        float p0[8], p1[8], alpha[8];
#pragma unroll
        for (int r = 0; r < 8; r++) {
            const int mrow = q0 + w * 16 + hoff + r;
            const int* mr  = mb + (size_t)mrow * SS + kv0;
            const int mv0  = mr[lcol];
            const int mv1  = mr[16 + lcol];
            float z0 = z[0][r] * 0.125f;
            float z1 = z[1][r] * 0.125f;
            z0 = mv0 ? z0 : -1e9f;
            z1 = mv1 ? z1 : -1e9f;
            float mt = fmaxf(z0, z1);
#pragma unroll
            for (int off = 1; off < 16; off <<= 1)
                mt = fmaxf(mt, __shfl_xor(mt, off, 32));
            const float mn = fmaxf(mi[r], mt);
            alpha[r] = __expf(mi[r] - mn);
            mi[r]    = mn;
            p0[r]    = __expf(z0 - mn);
            p1[r]    = __expf(z1 - mn);
            float rs = p0[r] + p1[r];
#pragma unroll
            for (int off = 1; off < 16; off <<= 1)
                rs += __shfl_xor(rs, off, 32);
            li[r] = li[r] * alpha[r] + rs;
        }

        // ---- transpose P into A-fragment layout via wave-private LDS ----
        // Same-wave LDS ops are architecturally in-order (DScnt), so no
        // barrier is needed between these stores and the fragment loads.
        _Float16* Pw = &Pl[w * 16 * LDP];
#pragma unroll
        for (int r = 0; r < 8; r++) {
            Pw[(hoff + r) * LDP + lcol]      = (_Float16)p0[r];
            Pw[(hoff + r) * LDP + 16 + lcol] = (_Float16)p1[r];
        }
        const _Float16* pr = &Pw[lcol * LDP];
        v16h pf = cat8(*(const v8h*)(pr + kbA), *(const v8h*)(pr + kbA + 8));

        // ---- acc = acc*alpha + P V: load all 4 V-fragments, then 4 WMMAs ----
        v16h vf[4];
#pragma unroll
        for (int nt = 0; nt < 4; nt++) {
            const _Float16* vp = &Vt[buf][(nt * 16 + lcol) * LDV + kbB];
            vf[nt] = cat8(*(const v8h*)vp, *(const v8h*)(vp + 8));
        }
#pragma unroll
        for (int nt = 0; nt < 4; nt++) {
#pragma unroll
            for (int r = 0; r < 8; r++) acc[nt][r] *= alpha[r];
            acc[nt] = wmma_f16(pf, vf[nt], acc[nt]);
        }
        // No end-of-loop barrier: double buffering + next iteration's
        // post-staging barrier order all cross-buffer hazards.
    }

    // ---- normalize + store concat [B*S, H*64] ----
#pragma unroll
    for (int nt = 0; nt < 4; nt++) {
#pragma unroll
        for (int r = 0; r < 8; r++) {
            const int row = q0 + w * 16 + hoff + r;
            const float o = acc[nt][r] / li[r];
            AO[((size_t)b * SS + row) * EE + h * DK + nt * 16 + lcol] =
                (_Float16)o;
        }
    }
}

// ---------------------------------------------------------------------------
extern "C" void kernel_launch(void* const* d_in, const int* in_sizes, int n_in,
                              void* d_out, int out_size, void* d_ws,
                              size_t ws_size, hipStream_t stream) {
    (void)in_sizes; (void)n_in; (void)out_size; (void)ws_size;
    const float* q    = (const float*)d_in[0];
    const float* k    = (const float*)d_in[1];
    const float* v    = (const float*)d_in[2];
    const int*   mask = (const int*)d_in[3];
    const float* Wq   = (const float*)d_in[4];
    const float* Wk   = (const float*)d_in[5];
    const float* Wv   = (const float*)d_in[6];
    const float* W    = (const float*)d_in[7];

    const size_t bufBytes = (size_t)MM * EE * sizeof(_Float16);  // 6.29 MB
    char* ws = (char*)d_ws;
    _Float16* Qh = (_Float16*)(ws + 0 * bufBytes);
    _Float16* Kh = (_Float16*)(ws + 1 * bufBytes);
    _Float16* Vh = (_Float16*)(ws + 2 * bufBytes);
    _Float16* AO = (_Float16*)(ws + 3 * bufBytes);

    dim3 gemmGrid(EE / 64, MM / 128);  // (12, 32)
    dim3 blk(256);

    // Projections: fp32 A, per-head B, f16 output
    gemm_wmma_kernel<1, true, false><<<gemmGrid, blk, 0, stream>>>(q, Wq, Qh);
    gemm_wmma_kernel<1, true, false><<<gemmGrid, blk, 0, stream>>>(k, Wk, Kh);
    gemm_wmma_kernel<1, true, false><<<gemmGrid, blk, 0, stream>>>(v, Wv, Vh);

    // Flash attention
    dim3 attnGrid(SS / 128, HH, BB);  // (16, 12, 2)
    attn_wmma_kernel<<<attnGrid, blk, 0, stream>>>(Qh, Kh, Vh, mask, AO);

    // Output projection: f16 A (async-staged), plain B, fp32 output
    gemm_wmma_kernel<0, false, true><<<gemmGrid, blk, 0, stream>>>(AO, W,
                                                                   d_out);
}
